// CoAttentionFusionBlock_62826781605920
// MI455X (gfx1250) — matrix operand order repaired and checked by reading, not compiled, not run on hardware
//
#include <hip/hip_runtime.h>

// out = [feat_depth | feat_rgb] @ W_f^T + b_f   (softmax over a singleton
// axis is exactly 1.0 -> the k1p/k2p/score branch is mathematically dead).
// M=32768, N=1024, K=2048. bf16 WMMA, f32 accumulate.
//
// Block tile 256(M) x 128(N), 8 wave32 arranged 4(M) x 2(N), 64x64 per wave
// -> per K-step: 16 ds_load_b128 : 16 v_wmma (1:1).

typedef __attribute__((ext_vector_type(16))) __bf16 v16bf;
typedef __attribute__((ext_vector_type(8)))  float  v8f;
typedef __attribute__((ext_vector_type(2)))  __bf16 v2bf;

union FragAB {
  v16bf v;
  uint4 u[2];
};

#if __has_builtin(__builtin_amdgcn_cvt_pk_bf16_f32)
__device__ __forceinline__ unsigned int pack2_bf16(float a, float b) {
  v2bf t = __builtin_amdgcn_cvt_pk_bf16_f32(a, b);
  unsigned int r;
  __builtin_memcpy(&r, &t, 4);
  return r;
}
#else
__device__ __forceinline__ unsigned int pack2_bf16(float a, float b) {
  // round-to-nearest-even f32 -> bf16, packed pair
  unsigned int ua = __float_as_uint(a);
  unsigned int ub = __float_as_uint(b);
  ua = (ua + 0x7FFFu + ((ua >> 16) & 1u)) >> 16;
  ub = (ub + 0x7FFFu + ((ub >> 16) & 1u)) >> 16;
  return (ua & 0xFFFFu) | (ub << 16);
}
#endif

#define TB_M 256
#define TB_N 128
#define TB_K 64
#define LDSS 72   // padded LDS row stride in bf16 elements (144B rows)
#define D_MODEL 1024
#define K_TOTAL 2048
#define NSTAGES (K_TOTAL / TB_K)  // 32

__global__ __launch_bounds__(256) void coattn_fused_gemm(
    const float* __restrict__ feat_rgb,
    const float* __restrict__ feat_depth,
    const float* __restrict__ W_f,
    const float* __restrict__ b_f,
    float* __restrict__ out) {
  __shared__ __align__(16) unsigned short As[TB_M * LDSS];
  __shared__ __align__(16) unsigned short Bs[TB_N * LDSS];

  const int tid  = threadIdx.x;
  const int lane = tid & 31;
  const int wave = tid >> 5;
  const int wm   = wave & 3;   // 4 waves tile M (64 rows each)
  const int wn   = wave >> 2;  // 2 waves tile N (64 cols each)
  const int N0   = blockIdx.x * TB_N;
  const int M0   = blockIdx.y * TB_M;

  const int lr = tid >> 4;         // 0..15 staging row in slab
  const int lc = (tid & 15) << 2;  // 0,4,...,60 staging col (floats)

  const int lhalf = (lane & 16) ? 1 : 0;
  const int l15   = lane & 15;

  // ---- accumulators with bias folded in (C/D layout: VGPR v -> M=v(+8),
  // N = lane&15; bias depends only on N -> splat per lane) ----
  v8f acc[4][4];
  {
    const int colb = N0 + 64 * wn + l15;
#pragma unroll
    for (int nt = 0; nt < 4; ++nt) {
      float bv = b_f[colb + 16 * nt];
#pragma unroll
      for (int mt = 0; mt < 4; ++mt)
#pragma unroll
        for (int v = 0; v < 8; ++v) acc[mt][nt][v] = bv;
    }
  }

  // ---- loop-invariant 32-bit per-thread BYTE offsets; the stage-varying
  // part lives in uniform scalar pointers -> SADDR-form global loads ----
  const unsigned offA = ((unsigned)(M0 + lr) * D_MODEL + (unsigned)lc) * 4u;
  const unsigned offB = ((unsigned)(N0 + lr) * K_TOTAL + (unsigned)lc) * 4u;

  float4 regA[16], regB[8];

  // prefetch K-stage 0 (k0 = 0 -> A source is feat_depth)
  {
    const char* pA = (const char*)feat_depth;
    const char* pB = (const char*)W_f;
#pragma unroll
    for (int i = 0; i < 16; ++i)
      regA[i] = *(const float4*)(pA + offA + (unsigned)(i * 16 * D_MODEL * 4));
#pragma unroll
    for (int i = 0; i < 8; ++i)
      regB[i] = *(const float4*)(pB + offB + (unsigned)(i * 16 * K_TOTAL * 4));
  }

  for (int s = 0; s < NSTAGES; ++s) {
    __syncthreads();  // previous compute done reading LDS
    // registers -> LDS with fp32->bf16 conversion
#pragma unroll
    for (int i = 0; i < 16; ++i) {
      uint2 p;
      p.x = pack2_bf16(regA[i].x, regA[i].y);
      p.y = pack2_bf16(regA[i].z, regA[i].w);
      *reinterpret_cast<uint2*>(&As[(lr + 16 * i) * LDSS + lc]) = p;
    }
#pragma unroll
    for (int i = 0; i < 8; ++i) {
      uint2 p;
      p.x = pack2_bf16(regB[i].x, regB[i].y);
      p.y = pack2_bf16(regB[i].z, regB[i].w);
      *reinterpret_cast<uint2*>(&Bs[(lr + 16 * i) * LDSS + lc]) = p;
    }
    __syncthreads();  // LDS tile ready

    // prefetch next stage (latency hidden behind WMMAs below)
    if (s + 1 < NSTAGES) {
      const int k0 = (s + 1) * TB_K;
      const char* pA = (const char*)((k0 < D_MODEL)
                                         ? feat_depth + k0
                                         : feat_rgb + (k0 - D_MODEL));
      const char* pB = (const char*)(W_f + k0);
#pragma unroll
      for (int i = 0; i < 16; ++i)
        regA[i] =
            *(const float4*)(pA + offA + (unsigned)(i * 16 * D_MODEL * 4));
#pragma unroll
      for (int i = 0; i < 8; ++i)
        regB[i] =
            *(const float4*)(pB + offB + (unsigned)(i * 16 * K_TOTAL * 4));
    }

    // ---- compute: two K=32 WMMA steps from this LDS stage ----
#pragma unroll
    for (int ks = 0; ks < TB_K; ks += 32) {
      FragAB a[4], b[4];
      // A (16x32 bf16): lanes 0-15 hold K 0-7 & 16-23 of row M=lane;
      // lanes 16-31 hold K 8-15 & 24-31.
      const int kA0 = ks + (lhalf ? 8 : 0);
#pragma unroll
      for (int mt = 0; mt < 4; ++mt) {
        int m = 64 * wm + 16 * mt + l15;
        a[mt].u[0] = *reinterpret_cast<const uint4*>(&As[m * LDSS + kA0]);
        a[mt].u[1] = *reinterpret_cast<const uint4*>(&As[m * LDSS + kA0 + 16]);
      }
      // B (32x16 bf16): lanes 0-15 hold K 0-7 & 16-23 of column N=lane;
      // lanes 16-31 hold K 8-15 & 24-31 (Bs rows are K-contiguous).
      const int kB0 = ks + (lhalf ? 8 : 0);
#pragma unroll
      for (int nt = 0; nt < 4; ++nt) {
        int n = 64 * wn + 16 * nt + l15;
        b[nt].u[0] = *reinterpret_cast<const uint4*>(&Bs[n * LDSS + kB0]);
        b[nt].u[1] = *reinterpret_cast<const uint4*>(&Bs[n * LDSS + kB0 + 16]);
      }
#pragma unroll
      for (int mt = 0; mt < 4; ++mt)
#pragma unroll
        for (int nt = 0; nt < 4; ++nt)
          acc[mt][nt] = __builtin_amdgcn_wmma_f32_16x16x32_bf16(
              false, a[mt].v, false, b[nt].v, (short)0, acc[mt][nt],
              false, false);
    }
  }

  // ---- epilogue: one 32-bit voffset, all tile steps as immediates ----
  {
    const char* outb = (const char*)out;
    const unsigned offO =
        ((unsigned)(M0 + 64 * wm + (lhalf ? 8 : 0)) * D_MODEL +
         (unsigned)(N0 + 64 * wn + l15)) * 4u;
#pragma unroll
    for (int mt = 0; mt < 4; ++mt)
#pragma unroll
      for (int nt = 0; nt < 4; ++nt)
#pragma unroll
        for (int v = 0; v < 8; ++v)
          *(float*)(outb + offO +
                    (unsigned)(mt * 16 * D_MODEL * 4 + v * D_MODEL * 4 +
                               nt * 16 * 4)) = acc[mt][nt][v];
  }
}

extern "C" void kernel_launch(void* const* d_in, const int* in_sizes, int n_in,
                              void* d_out, int out_size, void* d_ws,
                              size_t ws_size, hipStream_t stream) {
  (void)in_sizes; (void)n_in; (void)d_ws; (void)ws_size; (void)out_size;
  const float* feat_rgb   = (const float*)d_in[0];
  const float* feat_depth = (const float*)d_in[1];
  // d_in[2..5] = W_k1, b_k1, W_k2, b_k2: mathematically dead (softmax over a
  // length-1 axis is exactly 1.0), so never touched.
  const float* W_f = (const float*)d_in[6];
  const float* b_f = (const float*)d_in[7];
  float* out = (float*)d_out;

  dim3 grid(D_MODEL / TB_N, 32768 / TB_M);  // (8, 128)
  dim3 block(256);                           // 8 wave32 per workgroup
  hipLaunchKernelGGL(coattn_fused_gemm, grid, block, 0, stream,
                     feat_rgb, feat_depth, W_f, b_f, out);
}
